// SlayerNetwork_10316511445303
// MI455X (gfx1250) — matrix operand order repaired and checked by reading, not compiled, not run on hardware
//
#include <hip/hip_runtime.h>

// ---------------------------------------------------------------------------
// SLAYER SNN forward on MI455X (gfx1250, wave32, WMMA + TDM).
//
// Pipeline (per layer, sequenced on `stream`):
//   conv_wmma  : implicit-GEMM 3x3 conv, v_wmma_f32_16x16x32_bf16,
//                one (b,t) slice per block; input slice staged into LDS via
//                the Tensor Data Mover (tensor_load_to_lds + s_wait_tensorcnt).
//   psp_spike  : fused SRM(psp) + refractory spike recurrences over T,
//                all state in VGPRs, fused 2x2 avg pool, global_prefetch.
//   linear_wmma: readout as GEMM, N = 16 timesteps per tile, K = 1024,
//                time index clamped (no exec-masked loads in the k-loop).
//
// Activations are bf16 (spikes / pooled spikes are exact in bf16),
// accumulation & recurrences in f32.
//
// Workspace layout (requires ~131.2 MB):
//   [0)            actA : 6,553,600 ushort (bf16)
//   [13,107,200)   actB : 6,553,600 ushort
//   [26,214,400)   conv : 26,214,400 float
//   [131,072,000)  packed weights (bf16), ~84 KB
// ---------------------------------------------------------------------------

#define B_ 32
#define T_ 100

typedef __bf16 v16bf __attribute__((ext_vector_type(16)));
typedef float  v8f   __attribute__((ext_vector_type(8)));
typedef unsigned int u32x4 __attribute__((ext_vector_type(4)));
typedef int          i32x4 __attribute__((ext_vector_type(4)));
typedef int          i32x8 __attribute__((ext_vector_type(8)));

union Frag16 {
  v16bf v;
  unsigned short u[16];
  uint4 q[2];
};

__device__ __forceinline__ unsigned short f2bf(float f) {
  unsigned int u = __float_as_uint(f);
  unsigned int r = (u + 0x7FFFu + ((u >> 16) & 1u)) >> 16;  // RNE
  return (unsigned short)r;
}

// ---------------------------------------------------------------------------
// TDM: DMA `ndwords` dwords from global memory into LDS offset `lds_off`,
// expressed as a 1-row 2D tile per the D# layout (CDNA5 ISA).
// Issued per-wave (EXEC ignored); caller must s_wait_tensorcnt + barrier.
// ---------------------------------------------------------------------------
#if __has_builtin(__builtin_amdgcn_tensor_load_to_lds)
#define USE_TDM 1
__device__ __forceinline__ void tdm_load_1d(unsigned lds_off, const void* gptr,
                                            unsigned ndwords) {
  const unsigned long long ga = (unsigned long long)gptr;
  u32x4 g0;
  g0[0] = 1u;                                   // count=1, user-mode D#
  g0[1] = lds_off;                              // lds_addr (bytes)
  g0[2] = (unsigned)(ga & 0xFFFFFFFFu);         // global_addr[31:0]
  g0[3] = (unsigned)((ga >> 32) & 0x1FFFFFFu)   // global_addr[56:32]
          | (2u << 30);                         // type = 2 ("image")
  i32x8 g1;
  g1[0] = (int)(2u << 16);                      // data_size=2 (4B), mask=0
  g1[1] = (int)((ndwords & 0xFFFFu) << 16);     // tensor_dim0[15:0] @ bit48
  g1[2] = (int)((ndwords >> 16) | (1u << 16));  // tensor_dim0[31:16], tensor_dim1=1
  g1[3] = (int)((ndwords & 0xFFFFu) << 16);     // tile_dim0 = ndwords
  g1[4] = 1;                                    // tile_dim1=1, tile_dim2=0
  g1[5] = (int)ndwords;                         // tensor_dim0_stride[31:0]
  g1[6] = 0;                                    // stride0[47:32], stride1[15:0]
  g1[7] = 0;
  const i32x4 z4 = {0, 0, 0, 0};
#if __clang_major__ >= 23
  const i32x8 z8 = {0, 0, 0, 0, 0, 0, 0, 0};
  __builtin_amdgcn_tensor_load_to_lds(g0, g1, z4, z4, z8, 0);
#else
  __builtin_amdgcn_tensor_load_to_lds(g0, g1, z4, z4, 0);
#endif
}
#endif

// ---------------------------------------------------------------------------
// Weight prep: w = g * v / ||v||  ->  bf16 dense [coP][KP], zero padded.
// One block per (padded) output channel.
// ---------------------------------------------------------------------------
__global__ void prep_weights_kernel(const float* __restrict__ v,
                                    const float* __restrict__ g,
                                    unsigned short* __restrict__ Wn,
                                    int COUT, int KD, int KP) {
  const int co  = blockIdx.x;
  const int tid = threadIdx.x;
  __shared__ float red[64];
  float s = 0.f;
  if (co < COUT) {
    for (int k = tid; k < KD; k += 64) {
      float f = v[(size_t)co * KD + k];
      s += f * f;
    }
  }
  red[tid] = s;
  __syncthreads();
  for (int off = 32; off > 0; off >>= 1) {
    if (tid < off) red[tid] += red[tid + off];
    __syncthreads();
  }
  float inv = 0.f;
  if (co < COUT) inv = g[co] * rsqrtf(red[0]);
  for (int k = tid; k < KP; k += 64) {
    float w = (co < COUT && k < KD) ? v[(size_t)co * KD + k] * inv : 0.f;
    Wn[(size_t)co * KP + k] = f2bf(w);
  }
}

// ---------------------------------------------------------------------------
// Elementwise f32 -> bf16 (input spike tensor conversion).
// ---------------------------------------------------------------------------
__global__ void f32_to_bf16_kernel(const float* __restrict__ x,
                                   unsigned short* __restrict__ y, size_t n) {
  size_t i = (size_t)blockIdx.x * blockDim.x + threadIdx.x;
  if (i < n) y[i] = f2bf(x[i]);
}

// ---------------------------------------------------------------------------
// Implicit-GEMM 3x3 conv (pad=1), one (b,t) slice per block (grid = B*T).
// GEMM: M = COUT (tiled 16), N = H*W pixels (tiled 16), K = CIN*9 (pad 32).
// A (weights): pre-packed dense bf16 [coP][KP]; per lane two 16B loads.
//   A layout: lane = M row; elems 0..7 -> K = k0..k0+7, elems 8..15 -> +16,
//   with k0 = kt*32 + 8*(lane>=16).
// B (im2col): lane = N col; elems 0..7 -> K = k0..k0+7, elems 8..15 -> +16 (as A);
//   gathered from the LDS-staged input slice with pad-1 bounds checks.
// C/D: VGPR r -> M = r + 8*(lane>=16), col = lane&15.
// ---------------------------------------------------------------------------
template <int CIN, int COUT, int H, int W>
__global__ __launch_bounds__(128) void conv_wmma_kernel(
    const unsigned short* __restrict__ act,  // bf16 [B*T][CIN][H][W]
    const unsigned short* __restrict__ Wn,   // bf16 [coP][KP]
    float* __restrict__ out) {               // f32 [B*T][COUT][H][W]
  constexpr int KD = CIN * 9;
  constexpr int KT = (KD + 31) / 32;
  constexpr int KP = KT * 32;
  constexpr int NPIX = H * W;
  constexpr int NT = NPIX / 16;             // all layers divide evenly
  constexpr int MT = (COUT + 15) / 16;
  constexpr int SLICE_IN = CIN * H * W;
  constexpr int SLICE_OUT = COUT * H * W;

  __shared__ unsigned short sIn[SLICE_IN];  // only LDS object -> offset 0

  const int bt  = blockIdx.x;
  const int tid = threadIdx.x;
  const unsigned short* src = act + (size_t)bt * SLICE_IN;

#if defined(USE_TDM)
  if (tid < 32) {  // wave 0 issues the DMA for the whole slice
    tdm_load_1d(0u, (const void*)src, (unsigned)(SLICE_IN / 2));
    __builtin_amdgcn_s_wait_tensorcnt(0);
  }
  __syncthreads();
#else
  for (int i = tid; i < SLICE_IN; i += 128) sIn[i] = src[i];
  __syncthreads();
#endif

  const int wave   = tid >> 5;
  const int lane   = tid & 31;
  const int half   = lane >> 4;  // 0: lanes 0-15, 1: lanes 16-31
  const int lane16 = lane & 15;

  for (int tile = wave; tile < MT * NT; tile += 4) {
    const int mt = tile / NT;
    const int nt = tile % NT;

    const int pixel = nt * 16 + lane16;
    const int oh = pixel / W;
    const int ow = pixel % W;
    const int co_a = mt * 16 + lane16;

    v8f c = {};
    for (int kt = 0; kt < KT; ++kt) {
      // ---- A fragment: two contiguous 16B loads from packed weights ----
      Frag16 a;
      const unsigned short* wrow = Wn + (size_t)co_a * KP + kt * 32 + half * 8;
      a.q[0] = *(const uint4*)(wrow);
      a.q[1] = *(const uint4*)(wrow + 16);

      // ---- B fragment: im2col gather from LDS ----
      Frag16 b;
      const int kbase = kt * 32 + half * 8;
#pragma unroll
      for (int e = 0; e < 16; ++e) {
        const int k = kbase + e + ((e >> 3) << 3);
        unsigned short val = 0;
        if (k < KD) {
          const int ci = k / 9;
          const int r9 = k - ci * 9;
          const int kh = r9 / 3;
          const int kw = r9 - kh * 3;
          const int ih = oh - 1 + kh;
          const int iw = ow - 1 + kw;
          if (ih >= 0 && ih < H && iw >= 0 && iw < W)
            val = sIn[ci * (H * W) + ih * W + iw];
        }
        b.u[e] = val;
      }

      c = __builtin_amdgcn_wmma_f32_16x16x32_bf16(false, a.v, false, b.v,
                                                  (short)0, c, false, false);
    }

    float* dst = out + (size_t)bt * SLICE_OUT;
#pragma unroll
    for (int r = 0; r < 8; ++r) {
      const int co = mt * 16 + r + half * 8;
      if (co < COUT) dst[(size_t)co * NPIX + pixel] = c[r];
    }
  }
}

// ---------------------------------------------------------------------------
// Fused PSP (SRM alpha filter) + refractory spike dynamics over T, with
// optional fused 2x2 average pool. One thread per (pooled) output site;
// POOL=true keeps 4 sites of recurrence state (16 f32) in VGPRs.
// ---------------------------------------------------------------------------
template <bool POOL>
__global__ void psp_spike_kernel(const float* __restrict__ conv,    // [B][T][C][H][W]
                                 unsigned short* __restrict__ outa, // bf16 [B][T][C][H2][W2]
                                 int C, int H, int W) {
  const int H2 = POOL ? H / 2 : H;
  const int W2 = POOL ? W / 2 : W;
  const int total = B_ * C * H2 * W2;
  const int idx = blockIdx.x * blockDim.x + threadIdx.x;
  if (idx >= total) return;

  const int w2 = idx % W2;
  const int h2 = (idx / W2) % H2;
  const int c  = (idx / (W2 * H2)) % C;
  const int b  = idx / (W2 * H2 * C);

  // exp(-1/tau), (e/tau)*d, (-2*theta*scaleRef*e/tau)*d  (tauMem=tauRef=10)
  const float D   = 0.9048374180359596f;
  const float CD  = 0.2718281828459045f * D;
  const float CRD = -0.05436563656918090f * D;
  const float THETA = 0.1f;

  constexpr int NS = POOL ? 4 : 1;
  const size_t slice  = (size_t)C * H * W;
  const size_t oslice = (size_t)C * H2 * W2;

  int site[NS];
  if (POOL) {
#pragma unroll
    for (int i = 0; i < 4; ++i) {
      const int dh = i >> 1, dw = i & 1;
      site[i] = c * H * W + (2 * h2 + dh) * W + (2 * w2 + dw);
    }
  } else {
    site[0] = c * H * W + h2 * W + w2;
  }

  const float* base = conv + (size_t)b * T_ * slice;
  const size_t obase = (size_t)b * T_ * oslice + c * (H2 * W2) + h2 * W2 + w2;

  float pe1[NS], pr[NS], se1[NS], sr[NS];
#pragma unroll
  for (int i = 0; i < NS; ++i) { pe1[i] = 0.f; pr[i] = 0.f; se1[i] = 0.f; sr[i] = 0.f; }

  for (int t = 0; t < T_; ++t) {
    const float* pt = base + (size_t)t * slice;
    if (t + 1 < T_) __builtin_prefetch(pt + slice + site[0], 0, 1);  // global_prefetch_b8
    float acc = 0.f;
#pragma unroll
    for (int i = 0; i < NS; ++i) {
      const float x = pt[site[i]];
      pr[i] = D * pr[i] + CD * pe1[i];    // psp[t] (inputs strictly before t)
      sr[i] = D * sr[i] + CRD * se1[i];   // refractory response
      const float u = pr[i] + sr[i];
      const float s = (u >= THETA) ? 1.f : 0.f;
      se1[i] = D * se1[i] + s;
      pe1[i] = D * pe1[i] + x;
      acc += s;
    }
    outa[obase + (size_t)t * oslice] = f2bf(POOL ? acc * 0.25f : acc);
  }
}

// ---------------------------------------------------------------------------
// Readout: conv 4x4 (no pad) over the 4x4 map == GEMM with K = 64*16 = 1024.
// N dimension = 16 timesteps per tile (columns), M = 16 (11 padded).
// Time index clamped so all loads are unconditional (no exec masking in the
// k-loop); duplicate columns are computed but never stored.
// ---------------------------------------------------------------------------
__global__ __launch_bounds__(32) void linear_wmma_kernel(
    const unsigned short* __restrict__ act,  // bf16 [B][T][1024]
    const unsigned short* __restrict__ Wn,   // bf16 [16][1024]
    float* __restrict__ out) {               // f32 [B][T][11]
  constexpr int KD = 1024;
  constexpr int KT = KD / 32;

  const int b  = blockIdx.x;
  const int tt = blockIdx.y;
  const int lane   = threadIdx.x;
  const int half   = lane >> 4;
  const int lane16 = lane & 15;
  const int t  = tt * 16 + lane16;
  const int tc = (t < T_) ? t : (T_ - 1);  // clamp: loads always valid

  const unsigned short* arow = act + ((size_t)b * T_ + tc) * KD;

  v8f c = {};
  for (int kt = 0; kt < KT; ++kt) {
    Frag16 a;
    const unsigned short* wrow = Wn + (size_t)lane16 * KD + kt * 32 + half * 8;
    a.q[0] = *(const uint4*)(wrow);
    a.q[1] = *(const uint4*)(wrow + 16);

    Frag16 bf;
    const unsigned short* p = arow + kt * 32 + half * 8;
    bf.q[0] = *(const uint4*)(p);
    bf.q[1] = *(const uint4*)(p + 16);

    c = __builtin_amdgcn_wmma_f32_16x16x32_bf16(false, a.v, false, bf.v,
                                                (short)0, c, false, false);
  }

  if (t < T_) {
#pragma unroll
    for (int r = 0; r < 8; ++r) {
      const int co = r + half * 8;
      if (co < 11) out[((size_t)b * T_ + t) * 11 + co] = c[r];
    }
  }
}

// ---------------------------------------------------------------------------
// Host launcher.
// ---------------------------------------------------------------------------
extern "C" void kernel_launch(void* const* d_in, const int* in_sizes, int n_in,
                              void* d_out, int out_size, void* d_ws, size_t ws_size,
                              hipStream_t stream) {
  const float* x  = (const float*)d_in[0];
  const float* cv[4] = {(const float*)d_in[1], (const float*)d_in[3],
                        (const float*)d_in[5], (const float*)d_in[7]};
  const float* cg[4] = {(const float*)d_in[2], (const float*)d_in[4],
                        (const float*)d_in[6], (const float*)d_in[8]};
  const float* lv = (const float*)d_in[9];
  const float* lg = (const float*)d_in[10];

  // ---- workspace carve (needs ~131.2 MB) ----
  constexpr size_t ACT_ELEMS  = (size_t)B_ * T_ * 2 * 32 * 32;   // 6,553,600
  constexpr size_t CONV_ELEMS = (size_t)B_ * T_ * 8 * 32 * 32;   // 26,214,400
  char* ws = (char*)d_ws;
  unsigned short* actA = (unsigned short*)ws;
  unsigned short* actB = actA + ACT_ELEMS;
  float* convBuf = (float*)(ws + 2 * ACT_ELEMS * sizeof(unsigned short));
  unsigned short* wbase =
      (unsigned short*)(ws + 2 * ACT_ELEMS * sizeof(unsigned short) +
                        CONV_ELEMS * sizeof(float));
  unsigned short* W0 = wbase;              // 16 x 32
  unsigned short* W1 = W0 + 16 * 32;       // 16 x 96
  unsigned short* W2 = W1 + 16 * 96;       // 32 x 160
  unsigned short* W3 = W2 + 32 * 160;      // 64 x 288
  unsigned short* WL = W3 + 64 * 288;      // 16 x 1024

  // ---- weight norm + pack to bf16 ----
  prep_weights_kernel<<<16, 64, 0, stream>>>(cv[0], cg[0], W0, 8, 18, 32);
  prep_weights_kernel<<<16, 64, 0, stream>>>(cv[1], cg[1], W1, 16, 72, 96);
  prep_weights_kernel<<<32, 64, 0, stream>>>(cv[2], cg[2], W2, 32, 144, 160);
  prep_weights_kernel<<<64, 64, 0, stream>>>(cv[3], cg[3], W3, 64, 288, 288);
  prep_weights_kernel<<<16, 64, 0, stream>>>(lv, lg, WL, 11, 1024, 1024);

  // ---- input spikes f32 -> bf16 ----
  {
    size_t n = ACT_ELEMS;
    f32_to_bf16_kernel<<<(unsigned)((n + 255) / 256), 256, 0, stream>>>(x, actA, n);
  }

  const int BT = B_ * T_;

  // ---- layer 0: conv(2->8, 32x32) -> psp/spike -> pool -> 16x16 ----
  conv_wmma_kernel<2, 8, 32, 32><<<BT, 128, 0, stream>>>(actA, W0, convBuf);
  {
    int total = B_ * 8 * 16 * 16;
    psp_spike_kernel<true><<<(total + 255) / 256, 256, 0, stream>>>(convBuf, actB, 8, 32, 32);
  }

  // ---- layer 1: conv(8->16, 16x16) -> pool -> 8x8 ----
  conv_wmma_kernel<8, 16, 16, 16><<<BT, 128, 0, stream>>>(actB, W1, convBuf);
  {
    int total = B_ * 16 * 8 * 8;
    psp_spike_kernel<true><<<(total + 255) / 256, 256, 0, stream>>>(convBuf, actA, 16, 16, 16);
  }

  // ---- layer 2: conv(16->32, 8x8) -> pool -> 4x4 ----
  conv_wmma_kernel<16, 32, 8, 8><<<BT, 128, 0, stream>>>(actA, W2, convBuf);
  {
    int total = B_ * 32 * 4 * 4;
    psp_spike_kernel<true><<<(total + 255) / 256, 256, 0, stream>>>(convBuf, actB, 32, 8, 8);
  }

  // ---- layer 3: conv(32->64, 4x4), no pool ----
  conv_wmma_kernel<32, 64, 4, 4><<<BT, 128, 0, stream>>>(actB, W3, convBuf);
  {
    int total = B_ * 64 * 4 * 4;
    psp_spike_kernel<false><<<(total + 255) / 256, 256, 0, stream>>>(convBuf, actA, 64, 4, 4);
  }

  // ---- readout: [B][T][1024] x [16][1024]^T -> [B][T][11] ----
  {
    dim3 grid(B_, (T_ + 15) / 16);
    linear_wmma_kernel<<<grid, 32, 0, stream>>>(actA, WL, (float*)d_out);
  }
}
